// CliffordSharedSimplicialMPNN_2774548873296
// MI455X (gfx1250) — matrix operand order repaired and checked by reading, not compiled, not run on hardware
//
#include <hip/hip_runtime.h>
#include <hip/hip_bf16.h>
#include <math.h>

// ---------------------------------------------------------------------------
// Clifford Simplicial MPNN forward, MI455X (gfx1250, wave32).
// Channel-mixing mvlinear einsums -> V_WMMA_F32_16X16X32_F16 (f32 accum).
// One wave32 per 16-item tile; all tile intermediates live in LDS.
// Weights pre-packed to fragment-ordered f16 so B loads are unconditional.
// ---------------------------------------------------------------------------

#define NTOT   1600     // 64 graphs * 25 nodes
#define NEDGE  38400    // 64 * 25*24
#define NGRAPH 64
#define HID    28
#define TI     16       // items (edges/nodes) per tile
#define KMAX   64       // padded max input channels (62 for edge layer 0)
#define NOUT0  320      // 64 * 5 type-0 nodes

typedef __attribute__((ext_vector_type(16))) _Float16 v16h;
typedef __attribute__((ext_vector_type(8)))  _Float16 v8h;
typedef __attribute__((ext_vector_type(8)))  float    v8f;

// Blade order: 1, e1, e2, e3, e12, e13, e23, e123  (constexpr -> folded)
constexpr int G8c[8] = {0,1,1,1,2,2,2,3};
// Cayley product: blade(i)*blade(j) = STAB[i][j] * blade(KTAB[i][j])
constexpr int KTAB[8][8] = {
  {0,1,2,3,4,5,6,7},
  {1,0,4,5,2,3,7,6},
  {2,4,0,6,1,7,3,5},
  {3,5,6,0,7,1,2,4},
  {4,2,1,7,0,6,5,3},
  {5,3,7,1,6,0,4,2},
  {6,7,3,2,5,4,0,1},
  {7,6,5,4,3,2,1,0}};
constexpr float STAB[8][8] = {
  { 1, 1, 1, 1, 1, 1, 1, 1},
  { 1, 1, 1, 1, 1, 1, 1, 1},
  { 1,-1, 1, 1,-1,-1, 1,-1},
  { 1,-1,-1, 1, 1,-1,-1, 1},
  { 1,-1, 1, 1,-1,-1, 1,-1},
  { 1,-1,-1, 1, 1,-1,-1, 1},
  { 1, 1,-1, 1,-1, 1,-1,-1},
  { 1, 1,-1, 1,-1, 1,-1,-1}};
// (g1,g2,g3) -> index into the 20 nonzero grade-paths (np.nonzero(_P) order)
constexpr signed char P3LUT[64] = {
   0,-1,-1,-1,-1, 1,-1,-1,-1,-1, 2,-1,-1,-1,-1, 3,
  -1, 4,-1,-1, 5,-1, 6,-1,-1, 7,-1, 8,-1,-1, 9,-1,
  -1,-1,10,-1,-1,11,-1,12,13,-1,14,-1,-1,15,-1,-1,
  -1,-1,-1,16,-1,-1,17,-1,-1,18,-1,-1,19,-1,-1,-1};

struct BlkOff {
  // float offsets into params
  int lin_w, lin_b, silu_a, silu_b;
  int right_w, left_w, left_b, norm_a, sgp_w, ln_a;
  // half offsets into packed-weight region
  int pk_lin, pk_right, pk_left;
  int din, dout;
};
struct EmbOff {
  int sim, cl0_w, cl0_b, fe_w, fe_b;
  BlkOff cl1, cl2;
};

struct Smem {
  _Float16 A[TI][8][KMAX];   // f16 GEMM operand: [item][component][channel]
  float    X[TI][HID][8];    // post-silu activation (sgp input)
  float    XR[TI][HID][8];   // normalized right-branch
  float    Y[TI][HID][8];    // GEMM outputs / combine buffer
  float    red[TI];
};

__device__ __forceinline__ float sigm(float z) { return 1.f / (1.f + __expf(-z)); }

// ---------------------------------------------------------------------------
// WMMA GEMM over packed weights.
//   y[item][n][i] = sum_m A[item][i][m] * W[n][m][grade(i)]   (+ bias at i==0)
// Packed B layout: frag(g,nt,kt) = 32 lanes x 16 halves, contiguous.
// ---------------------------------------------------------------------------
template<int DIN, int DOUT>
__device__ __forceinline__ void mv_gemm(Smem& s, const _Float16* __restrict__ Wp,
                                        const float* __restrict__ bias, int lane) {
  constexpr int NT = (DOUT + 15) / 16;
  constexpr int KT = (DIN + 31) / 32;
  const int m  = lane & 15;
  const int hb = (lane & 16) ? 8 : 0;   // A K half-offset for hi lanes
  const int n  = (lane & 15);           // D/B column within tile
  const int mb = (lane & 16) ? 8 : 0;   // D row base
  #pragma unroll
  for (int comp = 0; comp < 8; ++comp) {
    const int g = G8c[comp];
    #pragma unroll
    for (int nt = 0; nt < NT; ++nt) {
      v8f acc = {};
      #pragma unroll
      for (int kt = 0; kt < KT; ++kt) {
        const _Float16* ap = &s.A[m][comp][kt * 32 + hb];   // 16B aligned
        v8h alo = *(const v8h*)(ap);
        v8h ahi = *(const v8h*)(ap + 16);
        v16h a = __builtin_shufflevector(alo, ahi,
                   0,1,2,3,4,5,6,7,8,9,10,11,12,13,14,15);
        v16h b = *(const v16h*)(Wp + (((((g * NT + nt) * KT + kt) * 32) + lane) << 4));
        acc = __builtin_amdgcn_wmma_f32_16x16x32_f16(
            /*neg_a=*/false, a, /*neg_b=*/false, b,
            /*c_mod=*/(short)0, acc, /*reuse_a=*/false, /*reuse_b=*/false);
      }
      const int nc = nt * 16 + n;
      const bool valid = (nc < DOUT);
      float bv = 0.f;
      if (bias) bv = bias[valid ? nc : 0];
      #pragma unroll
      for (int r = 0; r < 8; ++r) {
        float v = acc[r];
        if (comp == 0) v += bv;
        if (valid) s.Y[mb + r][nc][comp] = v;
      }
    }
  }
}

// ---------------------------------------------------------------------------
// One cemlp block (lin -> mvsilu -> sgp -> mvlayernorm) on a 16-item tile.
// Input: s.A (f16, DIN channels zero-padded to K-tile). Output: s.X + s.A
// (28 real channels, channels 28..31 of A zeroed for the next block).
// ---------------------------------------------------------------------------
template<int DIN>
__device__ void cemlp_block_tile(const float* __restrict__ P, const BlkOff& bo,
                                 const _Float16* __restrict__ Wpk, Smem& s, int lane) {
  constexpr int DOUT = HID;

  // mvlinear
  mv_gemm<DIN, DOUT>(s, Wpk + bo.pk_lin, P + bo.lin_b, lane);
  __syncthreads();

  // mvsilu: Y -> X (and f16 copy into A)
  for (int t = lane; t < TI * DOUT; t += 32) {
    const int item = t / DOUT, ch = t - item * DOUT;
    float xv[8];
    #pragma unroll
    for (int i = 0; i < 8; ++i) xv[i] = s.Y[item][ch][i];
    float invs[4];
    invs[0] = xv[0];
    invs[1] = xv[1]*xv[1] + xv[2]*xv[2] + xv[3]*xv[3];
    invs[2] = xv[4]*xv[4] + xv[5]*xv[5] + xv[6]*xv[6];
    invs[3] = xv[7]*xv[7];
    float gate[4];
    #pragma unroll
    for (int g = 0; g < 4; ++g)
      gate[g] = sigm(P[bo.silu_a + ch*4 + g] * invs[g] + P[bo.silu_b + ch*4 + g]);
    #pragma unroll
    for (int i = 0; i < 8; ++i) {
      float v = xv[i] * gate[G8c[i]];
      s.X[item][ch][i] = v;
      s.A[item][i][ch] = (_Float16)v;
    }
  }
  __syncthreads();

  // sgp right branch: X @ right -> Y
  mv_gemm<DOUT, DOUT>(s, Wpk + bo.pk_right, nullptr, lane);
  __syncthreads();

  // normalization: Y -> XR
  for (int t = lane; t < TI * DOUT; t += 32) {
    const int item = t / DOUT, ch = t - item * DOUT;
    float xv[8];
    #pragma unroll
    for (int i = 0; i < 8; ++i) xv[i] = s.Y[item][ch][i];
    float nn[4];
    nn[0] = sqrtf(xv[0]*xv[0] + 1e-12f);
    nn[1] = sqrtf(xv[1]*xv[1] + xv[2]*xv[2] + xv[3]*xv[3] + 1e-12f);
    nn[2] = sqrtf(xv[4]*xv[4] + xv[5]*xv[5] + xv[6]*xv[6] + 1e-12f);
    nn[3] = sqrtf(xv[7]*xv[7] + 1e-12f);
    float sg[4];
    #pragma unroll
    for (int g = 0; g < 4; ++g)
      sg[g] = sigm(P[bo.norm_a + ch*4 + g]) * (nn[g] - 1.f) + 1.f;
    #pragma unroll
    for (int i = 0; i < 8; ++i)
      s.XR[item][ch][i] = xv[i] / (sg[G8c[i]] + 1e-6f);
  }
  __syncthreads();

  // sgp left branch: X @ left + bias -> Y
  mv_gemm<DOUT, DOUT>(s, Wpk + bo.pk_left, P + bo.left_b, lane);
  __syncthreads();

  // quadratic geometric-product term, combine: Y = (Y + quad)/sqrt(2)
  for (int t = lane; t < TI * DOUT; t += 32) {
    const int item = t / DOUT, ch = t - item * DOUT;
    float w20[20];
    #pragma unroll
    for (int p = 0; p < 20; ++p) w20[p] = P[bo.sgp_w + ch*20 + p];
    float xr8[8], xx8[8];
    #pragma unroll
    for (int i = 0; i < 8; ++i) { xr8[i] = s.XR[item][ch][i]; xx8[i] = s.X[item][ch][i]; }
    float q[8] = {0,0,0,0,0,0,0,0};
    #pragma unroll
    for (int i = 0; i < 8; ++i) {
      #pragma unroll
      for (int j = 0; j < 8; ++j) {
        constexpr auto kf = KTAB;  // force constexpr context
        const int k  = kf[i][j];
        const int pi = P3LUT[(G8c[i] << 4) | (G8c[j] << 2) | G8c[k]];
        q[j] += xx8[i] * STAB[i][j] * w20[pi] * xr8[k];
      }
    }
    #pragma unroll
    for (int j = 0; j < 8; ++j)
      s.Y[item][ch][j] = (s.Y[item][ch][j] + q[j]) * 0.70710678118654752f;
  }
  __syncthreads();

  // mvlayernorm: per-item mean of per-channel multivector norms
  if (lane < TI) {
    float acc = 0.f;
    for (int ch = 0; ch < DOUT; ++ch) {
      float ss = 1e-12f;
      #pragma unroll
      for (int i = 0; i < 8; ++i) { float v = s.Y[lane][ch][i]; ss += v * v; }
      acc += sqrtf(ss);
    }
    s.red[lane] = acc / (float)DOUT;
  }
  __syncthreads();
  // write-out + zero-pad channels [28,32) of A for the next block's K-tile
  for (int t = lane; t < TI * 32; t += 32) {
    const int item = t >> 5, ch = t & 31;
    if (ch < DOUT) {
      float sc = P[bo.ln_a + ch] / (s.red[item] + 1e-6f);
      #pragma unroll
      for (int i = 0; i < 8; ++i) {
        float v = s.Y[item][ch][i] * sc;
        s.X[item][ch][i] = v;
        s.A[item][i][ch] = (_Float16)v;
      }
    } else {
      #pragma unroll
      for (int i = 0; i < 8; ++i) s.A[item][i][ch] = (_Float16)0.f;
    }
  }
  __syncthreads();
}

// ---------------------------------------------------------------------------
// Scalar single-item cemlp block (3 output channels) for node-type embeddings
// ---------------------------------------------------------------------------
__device__ void cemlp_scalar3(const float* __restrict__ P, const BlkOff bo,
                              const float (*xin)[8], float (*xout)[8]) {
  const int din = bo.din;
  float y[3][8], xv[3][8], xr[3][8];
  for (int n = 0; n < 3; ++n)
    for (int i = 0; i < 8; ++i) {
      int g = G8c[i];
      float acc = (i == 0) ? P[bo.lin_b + n] : 0.f;
      for (int m = 0; m < din; ++m) acc += xin[m][i] * P[bo.lin_w + (n*din + m)*4 + g];
      y[n][i] = acc;
    }
  for (int n = 0; n < 3; ++n) {
    float invs[4];
    invs[0] = y[n][0];
    invs[1] = y[n][1]*y[n][1] + y[n][2]*y[n][2] + y[n][3]*y[n][3];
    invs[2] = y[n][4]*y[n][4] + y[n][5]*y[n][5] + y[n][6]*y[n][6];
    invs[3] = y[n][7]*y[n][7];
    float gate[4];
    for (int g = 0; g < 4; ++g)
      gate[g] = sigm(P[bo.silu_a + n*4 + g] * invs[g] + P[bo.silu_b + n*4 + g]);
    for (int i = 0; i < 8; ++i) xv[n][i] = y[n][i] * gate[G8c[i]];
  }
  for (int n = 0; n < 3; ++n) {
    float yr[8];
    for (int i = 0; i < 8; ++i) {
      int g = G8c[i]; float acc = 0.f;
      for (int m = 0; m < 3; ++m) acc += xv[m][i] * P[bo.right_w + (n*3 + m)*4 + g];
      yr[i] = acc;
    }
    float nn[4];
    nn[0] = sqrtf(yr[0]*yr[0] + 1e-12f);
    nn[1] = sqrtf(yr[1]*yr[1] + yr[2]*yr[2] + yr[3]*yr[3] + 1e-12f);
    nn[2] = sqrtf(yr[4]*yr[4] + yr[5]*yr[5] + yr[6]*yr[6] + 1e-12f);
    nn[3] = sqrtf(yr[7]*yr[7] + 1e-12f);
    float sg[4];
    for (int g = 0; g < 4; ++g) sg[g] = sigm(P[bo.norm_a + n*4 + g]) * (nn[g] - 1.f) + 1.f;
    for (int i = 0; i < 8; ++i) xr[n][i] = yr[i] / (sg[G8c[i]] + 1e-6f);
  }
  for (int n = 0; n < 3; ++n) {
    const float* wsg = P + bo.sgp_w + n * 20;
    float o8[8];
    for (int i = 0; i < 8; ++i) {
      int g = G8c[i]; float acc = (i == 0) ? P[bo.left_b + n] : 0.f;
      for (int m = 0; m < 3; ++m) acc += xv[m][i] * P[bo.left_w + (n*3 + m)*4 + g];
      o8[i] = acc;
    }
    #pragma unroll
    for (int i = 0; i < 8; ++i) {
      float xi = xv[n][i];
      #pragma unroll
      for (int j = 0; j < 8; ++j) {
        const int k  = KTAB[i][j];
        const int pi = P3LUT[(G8c[i] << 4) | (G8c[j] << 2) | G8c[k]];
        o8[j] += xi * STAB[i][j] * wsg[pi] * xr[n][k];
      }
    }
    for (int j = 0; j < 8; ++j) y[n][j] = o8[j] * 0.70710678118654752f;
  }
  float mean = 0.f;
  for (int n = 0; n < 3; ++n) {
    float ss = 1e-12f;
    for (int i = 0; i < 8; ++i) ss += y[n][i]*y[n][i];
    mean += sqrtf(ss);
  }
  mean *= (1.f / 3.f);
  for (int n = 0; n < 3; ++n) {
    float sc = P[bo.ln_a + n] / (mean + 1e-6f);
    for (int i = 0; i < 8; ++i) xout[n][i] = y[n][i] * sc;
  }
}

// ---------------------------------------------------------------------------
// Kernels
// ---------------------------------------------------------------------------
__global__ void csmpnn_zero(float* p, int n) {
  int i = blockIdx.x * 256 + threadIdx.x;
  if (i < n) p[i] = 0.f;
}

// Pack one mvlinear weight matrix w[n][k][4] (f32) into fragment-ordered,
// zero-padded f16: frag(g,nt,kt) = 32 lanes x 16 halves, contiguous.
__global__ void csmpnn_pack(const float* __restrict__ P, int src, _Float16* __restrict__ Wp,
                            int dst, int din, int dout) {
  const int KT = (din + 31) / 32, NT = (dout + 15) / 16;
  const int total = 4 * NT * KT * 32 * 16;
  int i = blockIdx.x * 256 + threadIdx.x;
  if (i >= total) return;
  int h    = i & 15;
  int lane = (i >> 4) & 31;
  int r    = i >> 9;
  int kt   = r % KT;  r /= KT;
  int nt   = r % NT;
  int g    = r / NT;
  int n = nt * 16 + (lane & 15);
  int k = kt * 32 + ((lane & 16) ? 8 : 0) + (h < 8 ? h : 16 + (h - 8));
  float w = (n < dout && k < din) ? P[src + (n * din + k) * 4 + g] : 0.f;
  Wp[dst + i] = (_Float16)w;
}

__global__ void csmpnn_gstats(const float* __restrict__ loc, const int* __restrict__ nt,
                              const int* __restrict__ bv, float* gsum, float* gcnt) {
  int n = blockIdx.x * 256 + threadIdx.x;
  if (n >= NTOT) return;
  if (nt[n] == 0) {
    int b = bv[n];
    atomicAdd(&gsum[b*3 + 0], loc[n*3 + 0]);
    atomicAdd(&gsum[b*3 + 1], loc[n*3 + 1]);
    atomicAdd(&gsum[b*3 + 2], loc[n*3 + 2]);
    atomicAdd(&gcnt[b], 1.f);
  }
}

__global__ void csmpnn_embed(const float* __restrict__ P, EmbOff eo,
                             const float* __restrict__ loc, const float* __restrict__ vel,
                             const float* __restrict__ chg, const int* __restrict__ nt,
                             const int* __restrict__ xind, const int* __restrict__ bv,
                             const float* __restrict__ gsum, const float* __restrict__ gcnt,
                             float* __restrict__ x) {
  int n = blockIdx.x * 64 + threadIdx.x;
  if (n >= NTOT) return;
  int t = nt[n];
  float f[9][8];
  for (int c = 0; c < 9; ++c)
    for (int i = 0; i < 8; ++i) f[c][i] = 0.f;
  int np = t + 1;
  for (int r = 0; r < np; ++r) {
    int idx = xind[n*3 + r];
    int b = bv[idx];
    float ic = 1.f / gcnt[b];
    for (int d = 0; d < 3; ++d) {
      f[r][1 + d]      = loc[idx*3 + d] - gsum[b*3 + d] * ic;  // loc_mean
      f[np + r][1 + d] = vel[idx*3 + d];
    }
    f[2*np + r][0] = chg[idx];
  }
  float xa[3][8];
  if (t == 0) {
    for (int nn = 0; nn < 3; ++nn)
      for (int i = 0; i < 8; ++i) {
        float acc = (i == 0) ? P[eo.cl0_b + nn] : 0.f;
        for (int m = 0; m < 3; ++m) acc += f[m][i] * P[eo.cl0_w + nn*3 + m];
        xa[nn][i] = acc;
      }
  } else if (t == 1) {
    cemlp_scalar3(P, eo.cl1, f, xa);
  } else {
    cemlp_scalar3(P, eo.cl2, f, xa);
  }
  float xc[6][8];
  for (int c = 0; c < 3; ++c)
    for (int i = 0; i < 8; ++i) xc[c][i] = xa[c][i];
  for (int c = 0; c < 3; ++c) {
    for (int i = 0; i < 8; ++i) xc[3 + c][i] = 0.f;
    xc[3 + c][0] = P[eo.sim + t*3 + c];   // sim_inv
  }
  for (int nn = 0; nn < HID; ++nn)
    for (int i = 0; i < 8; ++i) {
      float acc = (i == 0) ? P[eo.fe_b + nn] : 0.f;
      for (int m = 0; m < 6; ++m) acc += xc[m][i] * P[eo.fe_w + nn*6 + m];
      x[(n*HID + nn)*8 + i] = acc;
    }
}

__global__ __launch_bounds__(32)
void csmpnn_edge(const float* __restrict__ P, const _Float16* __restrict__ Wpk,
                 BlkOff b0, BlkOff b1, int sim_off,
                 const float* __restrict__ x, const int* __restrict__ eidx,
                 const int* __restrict__ eat, float* __restrict__ agg) {
  __shared__ Smem s;
  __shared__ int rcv[TI], snd[TI];
  const int lane = threadIdx.x;
  const int e0 = blockIdx.x * TI;
  __builtin_prefetch(Wpk + b0.pk_lin, 0, 1);   // global_prefetch_b8 on weights
  if (lane < TI) {
    rcv[lane] = eidx[e0 + lane];
    snd[lane] = eidx[NEDGE + e0 + lane];
  }
  __syncthreads();
  // m_in = [h[recv] (28) | h[send] (28) | edge_attr (6) | zero-pad to 64]
  for (int t = lane; t < TI * 64; t += 32) {
    const int item = t >> 6, c = t & 63;
    const int e = e0 + item;
    float v[8];
    #pragma unroll
    for (int i = 0; i < 8; ++i) v[i] = 0.f;
    if (c < 28) {
      const float* src = &x[(rcv[item]*HID + c)*8];
      #pragma unroll
      for (int i = 0; i < 8; ++i) v[i] = src[i];
    } else if (c < 56) {
      const float* src = &x[(snd[item]*HID + (c - 28))*8];
      #pragma unroll
      for (int i = 0; i < 8; ++i) v[i] = src[i];
    } else if (c < 62) {
      int cc = c - 56;
      int ty = eat[e*2 + (cc >= 3 ? 1 : 0)];
      v[0] = P[sim_off + ty*3 + (cc % 3)];
    }
    #pragma unroll
    for (int i = 0; i < 8; ++i) s.A[item][i][c] = (_Float16)v[i];
  }
  __syncthreads();
  cemlp_block_tile<62>(P, b0, Wpk, s, lane);
  cemlp_block_tile<28>(P, b1, Wpk, s, lane);
  // segment mean: every node receives exactly 24 edges
  for (int t = lane; t < TI * HID; t += 32) {
    const int item = t / HID, c = t - item * HID;
    float* dst = &agg[(rcv[item]*HID + c)*8];
    #pragma unroll
    for (int i = 0; i < 8; ++i)
      atomicAdd(&dst[i], s.X[item][c][i] * (1.f / 24.f));
  }
}

__global__ __launch_bounds__(32)
void csmpnn_node(const float* __restrict__ P, const _Float16* __restrict__ Wpk,
                 BlkOff b0, BlkOff b1, int sim_off,
                 float* __restrict__ x, const float* __restrict__ agg,
                 const int* __restrict__ nt) {
  __shared__ Smem s;
  const int lane = threadIdx.x;
  const int n0 = blockIdx.x * TI;
  // u_in = [h (28) | agg (28) | sim_inv (3) | zero-pad to 64]
  for (int t = lane; t < TI * 64; t += 32) {
    const int item = t >> 6, c = t & 63;
    const int n = n0 + item;
    float v[8];
    #pragma unroll
    for (int i = 0; i < 8; ++i) v[i] = 0.f;
    if (c < 28) {
      const float* src = &x[(n*HID + c)*8];
      #pragma unroll
      for (int i = 0; i < 8; ++i) v[i] = src[i];
    } else if (c < 56) {
      const float* src = &agg[(n*HID + (c - 28))*8];
      #pragma unroll
      for (int i = 0; i < 8; ++i) v[i] = src[i];
    } else if (c < 59) {
      v[0] = P[sim_off + nt[n]*3 + (c - 56)];
    }
    #pragma unroll
    for (int i = 0; i < 8; ++i) s.A[item][i][c] = (_Float16)v[i];
  }
  __syncthreads();
  cemlp_block_tile<59>(P, b0, Wpk, s, lane);
  cemlp_block_tile<28>(P, b1, Wpk, s, lane);
  // residual: h = h + cemlp(u_in)
  for (int t = lane; t < TI * HID; t += 32) {
    const int item = t / HID, c = t - item * HID;
    float* dst = &x[((n0 + item)*HID + c)*8];
    #pragma unroll
    for (int i = 0; i < 8; ++i) dst[i] += s.X[item][c][i];
  }
}

__global__ __launch_bounds__(32)
void csmpnn_out(const float* __restrict__ P, const _Float16* __restrict__ Wpk,
                BlkOff bproj, int pl_w, int pl_b,
                const float* __restrict__ x, const float* __restrict__ loc,
                const int* __restrict__ n0i, float* __restrict__ out) {
  __shared__ Smem s;
  __shared__ int nod[TI];
  const int lane = threadIdx.x;
  const int o0 = blockIdx.x * TI;
  if (lane < TI) nod[lane] = n0i[o0 + lane];
  __syncthreads();
  for (int t = lane; t < TI * 32; t += 32) {
    const int item = t >> 5, c = t & 31;
    float v[8];
    #pragma unroll
    for (int i = 0; i < 8; ++i) v[i] = 0.f;
    if (c < HID) {
      const float* src = &x[(nod[item]*HID + c)*8];
      #pragma unroll
      for (int i = 0; i < 8; ++i) v[i] = src[i];
    }
    #pragma unroll
    for (int i = 0; i < 8; ++i) s.A[item][i][c] = (_Float16)v[i];
  }
  __syncthreads();
  cemlp_block_tile<28>(P, bproj, Wpk, s, lane);
  if (lane < TI) {
    int o = o0 + lane;
    int node = nod[lane];
    float mv[8];
    #pragma unroll
    for (int i = 0; i < 8; ++i) {
      float acc = (i == 0) ? P[pl_b] : 0.f;
      for (int m = 0; m < HID; ++m) acc += s.X[lane][m][i] * P[pl_w + m*4 + G8c[i]];
      mv[i] = acc;
    }
    for (int d = 0; d < 3; ++d)
      out[o*3 + d] = loc[node*3 + d] + mv[1 + d];
  }
}

// ---------------------------------------------------------------------------
// Host side: parameter offsets (init_params insertion order, flat f32 in
// d_in[0]), packed-weight offsets, launch sequence.
// ---------------------------------------------------------------------------
static int pack_size(int din, int dout) {
  return 4 * ((dout + 15) / 16) * ((din + 31) / 32) * 32 * 16;   // halves
}
static BlkOff make_blk(int& off, int& pk, int a, int b) {
  BlkOff o; o.din = a; o.dout = b;
  o.lin_w  = off; off += b * a * 4;
  o.lin_b  = off; off += b;
  o.silu_a = off; off += b * 4;
  o.silu_b = off; off += b * 4;
  o.right_w = off; off += b * b * 4;     // right: no bias
  o.left_w  = off; off += b * b * 4;
  o.left_b  = off; off += b;
  o.norm_a  = off; off += b * 4;
  o.sgp_w   = off; off += b * 20;        // N_PATHS = 20
  o.ln_a    = off; off += b;
  o.pk_lin   = pk; pk += pack_size(a, b);
  o.pk_right = pk; pk += pack_size(b, b);
  o.pk_left  = pk; pk += pack_size(b, b);
  return o;
}

extern "C" void kernel_launch(void* const* d_in, const int* in_sizes, int n_in,
                              void* d_out, int out_size, void* d_ws, size_t ws_size,
                              hipStream_t stream) {
  (void)in_sizes; (void)n_in; (void)out_size; (void)ws_size;
  const float* P    = (const float*)d_in[0];
  const float* loc  = (const float*)d_in[1];
  const float* vel  = (const float*)d_in[2];
  const float* chg  = (const float*)d_in[3];
  const int* ntypes = (const int*)d_in[4];
  const int* xind   = (const int*)d_in[5];
  const int* bvec   = (const int*)d_in[6];
  const int* eidx   = (const int*)d_in[7];
  const int* eat    = (const int*)d_in[8];
  const int* n0i    = (const int*)d_in[9];
  float* out = (float*)d_out;

  float* ws   = (float*)d_ws;
  float* x    = ws;                        // NTOT*HID*8
  float* agg  = x + NTOT*HID*8;            // NTOT*HID*8
  float* gsum = agg + NTOT*HID*8;          // NGRAPH*3
  float* gcnt = gsum + NGRAPH*3;           // NGRAPH
  _Float16* Wpk = (_Float16*)(gcnt + NGRAPH);   // packed weights (32B aligned)

  // --- parameter offsets (init_params insertion order) ---
  int off = 0, pk = 0;
  EmbOff eo;
  eo.sim   = off; off += 9;                // sim_emb (3,3)
  eo.cl0_w = off; off += 9;                // cl_emb0 w (3,3)
  eo.cl0_b = off; off += 3;
  {
    int pk_dummy = 0;
    eo.cl1 = make_blk(off, pk_dummy, 6, 3);  // cl_emb1: one block 6->3 (scalar path)
    eo.cl2 = make_blk(off, pk_dummy, 9, 3);  // cl_emb2: one block 9->3 (scalar path)
  }
  eo.fe_w = off; off += HID * 6;           // feat_emb w (28,6)
  eo.fe_b = off; off += HID;
  BlkOff eblk[3][2], nblk[3][2];
  for (int l = 0; l < 3; ++l) {
    eblk[l][0] = make_blk(off, pk, 2*HID + 6, HID);  // edge MLP: 62->28
    eblk[l][1] = make_blk(off, pk, HID, HID);        //           28->28
    nblk[l][0] = make_blk(off, pk, 2*HID + 3, HID);  // node MLP: 59->28
    nblk[l][1] = make_blk(off, pk, HID, HID);        //           28->28
  }
  BlkOff proj = make_blk(off, pk, HID, HID);   // proj_mlp: 28->28
  int pl_w = off; off += HID * 4;          // proj_lin w (1,28,4)
  int pl_b = off; off += 1;

  // --- pack all WMMA weight matrices to fragment-ordered f16 ---
  BlkOff* allblk[13];
  int nb = 0;
  for (int l = 0; l < 3; ++l) {
    allblk[nb++] = &eblk[l][0]; allblk[nb++] = &eblk[l][1];
    allblk[nb++] = &nblk[l][0]; allblk[nb++] = &nblk[l][1];
  }
  allblk[nb++] = &proj;
  for (int bI = 0; bI < nb; ++bI) {
    const BlkOff* b = allblk[bI];
    int szl = pack_size(b->din, b->dout);
    int szs = pack_size(b->dout, b->dout);
    hipLaunchKernelGGL(csmpnn_pack, dim3((szl + 255)/256), dim3(256), 0, stream,
                       P, b->lin_w, Wpk, b->pk_lin, b->din, b->dout);
    hipLaunchKernelGGL(csmpnn_pack, dim3((szs + 255)/256), dim3(256), 0, stream,
                       P, b->right_w, Wpk, b->pk_right, b->dout, b->dout);
    hipLaunchKernelGGL(csmpnn_pack, dim3((szs + 255)/256), dim3(256), 0, stream,
                       P, b->left_w, Wpk, b->pk_left, b->dout, b->dout);
  }

  // --- forward pass ---
  hipLaunchKernelGGL(csmpnn_zero, dim3(1), dim3(256), 0, stream,
                     gsum, NGRAPH*3 + NGRAPH);
  hipLaunchKernelGGL(csmpnn_gstats, dim3((NTOT + 255)/256), dim3(256), 0, stream,
                     loc, ntypes, bvec, gsum, gcnt);
  hipLaunchKernelGGL(csmpnn_embed, dim3((NTOT + 63)/64), dim3(64), 0, stream,
                     P, eo, loc, vel, chg, ntypes, xind, bvec, gsum, gcnt, x);
  const int aggN = NTOT * HID * 8;
  for (int l = 0; l < 3; ++l) {
    hipLaunchKernelGGL(csmpnn_zero, dim3((aggN + 255)/256), dim3(256), 0, stream,
                       agg, aggN);
    hipLaunchKernelGGL(csmpnn_edge, dim3(NEDGE / TI), dim3(32), 0, stream,
                       P, Wpk, eblk[l][0], eblk[l][1], eo.sim, x, eidx, eat, agg);
    hipLaunchKernelGGL(csmpnn_node, dim3(NTOT / TI), dim3(32), 0, stream,
                       P, Wpk, nblk[l][0], nblk[l][1], eo.sim, x, agg, ntypes);
  }
  hipLaunchKernelGGL(csmpnn_out, dim3(NOUT0 / TI), dim3(32), 0, stream,
                     P, Wpk, proj, pl_w, pl_b, x, loc, n0i, out);
}
